// DotProductAttention_9543417332303
// MI455X (gfx1250) — compile-verified, matrix operand order checked by reading, not run on hardware
//
#include <hip/hip_runtime.h>
#include <hip/hip_bf16.h>

typedef _Float16 v16h __attribute__((ext_vector_type(16)));
typedef _Float16 h8   __attribute__((ext_vector_type(8)));
typedef _Float16 h2   __attribute__((ext_vector_type(2)));
typedef float    v8f  __attribute__((ext_vector_type(8)));

#define S_LEN   4096
#define D_DIM   64
#define KBLK    32            // keys per inner iteration
#define NBLK    (S_LEN / KBLK)
#define KSTRIDE 72            // halves; 144B row stride (16B aligned, bank-friendly)
#define VSTRIDE 40            // halves; 80B row stride (16B aligned, bank-friendly)
#define LOG2E   1.4426950408889634f

__device__ inline v8f splat8(float x) {
    v8f r;
#pragma unroll
    for (int e = 0; e < 8; ++e) r[e] = x;
    return r;
}

// Fragment element mapping (CDNA5 16-bit A/B layout):
//   element e in [0,8)  <-> row[doff + half*8 + e]
//   element e in [8,16) <-> row[doff + 16 + half*8 + (e-8)]
__device__ inline v16h lds_frag(const _Float16* row, int half, int doff) {
    union { v16h v; h8 h[2]; } u;
    u.h[0] = *reinterpret_cast<const h8*>(row + doff + half * 8);
    u.h[1] = *reinterpret_cast<const h8*>(row + doff + 16 + half * 8);
    return u.v;
}

// Global f32 -> f16 fragment with pre-scale folded in.
__device__ inline v16h glb_frag_scaled(const float* __restrict__ row, int half,
                                       int doff, float scale) {
    const float4* p0 = reinterpret_cast<const float4*>(row + doff + half * 8);
    const float4* p1 = reinterpret_cast<const float4*>(row + doff + 16 + half * 8);
    float4 a = p0[0], b = p0[1], c = p1[0], d = p1[1];
    float lo[8] = {a.x, a.y, a.z, a.w, b.x, b.y, b.z, b.w};
    float hi[8] = {c.x, c.y, c.z, c.w, d.x, d.y, d.z, d.w};
    v16h r;
#pragma unroll
    for (int e = 0; e < 8; ++e) {
        r[e]     = (_Float16)(lo[e] * scale);
        r[e + 8] = (_Float16)(hi[e] * scale);
    }
    return r;
}

__device__ inline v8f wmma_f16(v16h a, v16h b, v8f c) {
    return __builtin_amdgcn_wmma_f32_16x16x32_f16(false, a, false, b, (short)0, c, false, false);
}

// Cooperative stage of one 32x64 K block (row-major f16) and V block
// (d-major / transposed f16) into LDS. 128 threads, coalesced reads,
// packed ds_store_b32 writes.
__device__ inline void stage_kv(const float* __restrict__ Kp, const float* __restrict__ Vp,
                                _Float16* kt, _Float16* vt, int t) {
#pragma unroll
    for (int j = 0; j < 8; ++j) {
        int i = (t + j * 128) * 2;            // flat k*64 + d, d even
        int k = i >> 6;
        int d = i & 63;
        float2 kv = *reinterpret_cast<const float2*>(Kp + (size_t)k * D_DIM + d);
        h2 p; p[0] = (_Float16)kv.x; p[1] = (_Float16)kv.y;
        *reinterpret_cast<h2*>(kt + k * KSTRIDE + d) = p;    // row-major
    }
#pragma unroll
    for (int j = 0; j < 8; ++j) {
        int pi = t + j * 128;                 // pair index
        int k  = (pi >> 6) << 1;              // even k
        int d  = pi & 63;
        float a = Vp[(size_t)k * D_DIM + d];
        float b = Vp[(size_t)(k + 1) * D_DIM + d];
        h2 p; p[0] = (_Float16)a; p[1] = (_Float16)b;
        *reinterpret_cast<h2*>(vt + d * VSTRIDE + k) = p;    // transposed
    }
}

__global__ __launch_bounds__(128) void
flash_attn_fwd(const float* __restrict__ Q, const float* __restrict__ K,
               const float* __restrict__ V, const float* __restrict__ Mk,
               float* __restrict__ O) {
    const int lane = threadIdx.x & 31;
    const int wave = threadIdx.x >> 5;
    const int n    = lane & 15;   // query column of all transposed tiles
    const int half = lane >> 4;   // K-half of A/B fragments held by this lane

    const int bh   = blockIdx.x >> 5;        // 16 (b,h) slices
    const int qblk = blockIdx.x & 31;        // 32 query blocks of 128 rows
    const int q0   = qblk * 128 + wave * 32; // 32 queries per wave: two 16-row tiles

    const size_t base = (size_t)bh * S_LEN * D_DIM;
    const int qA = q0 + n, qB = q0 + 16 + n;
    const float* mrowA = Mk + (size_t)qA * S_LEN;   // mask broadcasts over B,H
    const float* mrowB = Mk + (size_t)qB * S_LEN;

    __shared__ _Float16 ktile[2][KBLK * KSTRIDE];   // 2 x 4.5 KB
    __shared__ _Float16 vtile[2][D_DIM * VSTRIDE];  // 2 x 5.0 KB

    // Loop-invariant Q^T B-fragments, pre-scaled by (1/sqrt(d_k))*log2(e):
    // scores come out of the QK WMMAs already in log2 domain.
    const float qs = 0.125f * LOG2E;
    v16h bq[2][2];
    bq[0][0] = glb_frag_scaled(Q + base + (size_t)qA * D_DIM, half, 0,  qs);
    bq[0][1] = glb_frag_scaled(Q + base + (size_t)qA * D_DIM, half, 32, qs);
    bq[1][0] = glb_frag_scaled(Q + base + (size_t)qB * D_DIM, half, 0,  qs);
    bq[1][1] = glb_frag_scaled(Q + base + (size_t)qB * D_DIM, half, 32, qs);

    v8f ot[2][4];                                   // O^T accumulators [tile][d-chunk]
#pragma unroll
    for (int t = 0; t < 2; ++t)
#pragma unroll
        for (int c = 0; c < 4; ++c) ot[t][c] = splat8(0.0f);

    float m_run[2] = {-INFINITY, -INFINITY};        // log2 domain
    float l_run[2] = {0.0f, 0.0f};
    const v8f mcoef = splat8(-1e9f * LOG2E);

    stage_kv(K + base, V + base, ktile[0], vtile[0], threadIdx.x);

    for (int kb = 0; kb < NBLK; ++kb) {
        __syncthreads();
        const int cur = kb & 1;
        if (kb + 1 < NBLK) {                        // overlap next block's staging
            const size_t nk0 = (size_t)(kb + 1) * KBLK * D_DIM;
            stage_kv(K + base + nk0, V + base + nk0,
                     ktile[1 - cur], vtile[1 - cur], threadIdx.x);
        }

        const int k0 = kb * KBLK;
        const _Float16* kt = ktile[cur];
        const _Float16* vt = vtile[cur];

        // ---- K A-fragments: loaded once, shared by both query tiles ----
        v16h ak00 = lds_frag(kt + n * KSTRIDE, half, 0);
        v16h ak01 = lds_frag(kt + n * KSTRIDE, half, 32);
        v16h ak10 = lds_frag(kt + (16 + n) * KSTRIDE, half, 0);
        v16h ak11 = lds_frag(kt + (16 + n) * KSTRIDE, half, 32);

        // ---- S^T = K_blk x Q^T : 8 WMMAs, two independent accumulator sets ----
        v8f st[2][2];
#pragma unroll
        for (int t = 0; t < 2; ++t) {
            v8f z0 = splat8(0.0f);
            v8f z1 = splat8(0.0f);
            z0 = wmma_f16(ak00, bq[t][0], z0);
            z1 = wmma_f16(ak10, bq[t][0], z1);
            z0 = wmma_f16(ak01, bq[t][1], z0);
            z1 = wmma_f16(ak11, bq[t][1], z1);
            st[t][0] = z0; st[t][1] = z1;
        }

        // ---- per-tile: mask + online softmax (base-2) + P^T fragment ----
        v16h bp[2];
#pragma unroll
        for (int t = 0; t < 2; ++t) {
            const float* mrow = t ? mrowB : mrowA;
            // 32B-aligned vector loads straight into v8f (2x global_load_b128)
            v8f mk0 = *reinterpret_cast<const v8f*>(mrow + k0 + half * 8);
            v8f mk1 = *reinterpret_cast<const v8f*>(mrow + k0 + 16 + half * 8);

            // whole-vector math: operands are already pk-pair aligned
            v8f s0 = mk0 * mcoef + st[t][0];
            v8f s1 = mk1 * mcoef + st[t][1];

            float mblk = -INFINITY;
#pragma unroll
            for (int v = 0; v < 8; ++v) mblk = fmaxf(mblk, fmaxf(s0[v], s1[v]));
            mblk = fmaxf(mblk, __shfl_xor(mblk, 16, 32));

            const float m_new = fmaxf(m_run[t], mblk);
            const float alpha = __builtin_amdgcn_exp2f(m_run[t] - m_new);

            v8f p0, p1;
            float rsum = 0.0f;
#pragma unroll
            for (int v = 0; v < 8; ++v) {
                p0[v] = __builtin_amdgcn_exp2f(s0[v] - m_new);
                p1[v] = __builtin_amdgcn_exp2f(s1[v] - m_new);
                rsum += p0[v] + p1[v];
            }
            rsum += __shfl_xor(rsum, 16, 32);
            l_run[t] = l_run[t] * alpha + rsum;
            m_run[t] = m_new;

            const v8f av_ = splat8(alpha);
#pragma unroll
            for (int c = 0; c < 4; ++c) ot[t][c] = ot[t][c] * av_;

#pragma unroll
            for (int v = 0; v < 8; ++v) {
                bp[t][v]     = (_Float16)p0[v];
                bp[t][v + 8] = (_Float16)p1[v];
            }
        }

        // ---- O^T += V^T x P^T : 8 WMMAs, V^T A-fragments shared by both tiles ----
#pragma unroll
        for (int c = 0; c < 4; ++c) {
            v16h av = lds_frag(vt + (c * 16 + n) * VSTRIDE, half, 0);
            ot[0][c] = wmma_f16(av, bp[0], ot[0][c]);
            ot[1][c] = wmma_f16(av, bp[1], ot[1][c]);
        }
    }

    // ---- epilogue: O[q][d] = O^T / l ; lane owns row q, d = c*16 + half*8 + v ----
#pragma unroll
    for (int t = 0; t < 2; ++t) {
        const v8f inv_l = splat8(1.0f / l_run[t]);
        float* orow = O + base + (size_t)(t ? qB : qA) * D_DIM;
#pragma unroll
        for (int c = 0; c < 4; ++c) {
            v8f ov = ot[t][c] * inv_l;
            *reinterpret_cast<v8f*>(orow + c * 16 + half * 8) = ov;  // 2x b128 store
        }
    }
}

extern "C" void kernel_launch(void* const* d_in, const int* in_sizes, int n_in,
                              void* d_out, int out_size, void* d_ws, size_t ws_size,
                              hipStream_t stream) {
    (void)in_sizes; (void)n_in; (void)d_ws; (void)ws_size; (void)out_size;
    const float* Q  = (const float*)d_in[0];
    const float* K  = (const float*)d_in[1];
    const float* V  = (const float*)d_in[2];
    // d_in[3] = d_k (scalar, 64) — folded into the Q pre-scale
    const float* Mk = (const float*)d_in[4];
    float* O = (float*)d_out;

    // grid: 16 (b,h) slices x 32 query-blocks of 128 rows; 128 threads = 4 waves
    dim3 grid(16 * 32), block(128);
    flash_attn_fwd<<<grid, block, 0, stream>>>(Q, K, V, Mk, O);
}
